// RelationalMemory_9981503996169
// MI455X (gfx1250) — matrix operand order repaired and checked by reading, not compiled, not run on hardware
//
#include <hip/hip_runtime.h>
#include <hip/hip_bf16.h>
#include <math.h>

// ---------------------------------------------------------------------------
// RelationalMemory forward for MI455X (gfx1250, wave32, WMMA).
// bf16 WMMA GEMMs (v_wmma_f32_16x16x32_bf16), f32 accumulate. Weights are
// transposed+converted to bf16 once per launch and stay hot in the 192MB L2
// across the 128-step scan. Each wave register-blocks a 32x32 C tile
// (2x2 WMMA tiles) so A/B fragments are each reused twice: 2 b128 loads per
// WMMA instead of 4. Attention is one wave per (b,h) with shuffle reductions
// (no LDS, no barriers). Host loop issues 9 kernels per step; stream order
// carries the recurrence dependency.
// ---------------------------------------------------------------------------

typedef __bf16 bf16;
typedef __attribute__((ext_vector_type(16))) __bf16 v16bf;
typedef __attribute__((ext_vector_type(8)))  __bf16 v8bf;
typedef __attribute__((ext_vector_type(8)))  float  v8f;

#define RB  64
#define RT  128
#define RS  3
#define RD  1024
#define RH  8
#define RDK 128

// ---------------------------------------------------------------------------
// Weight transpose + f32 -> bf16 convert.  Dest Wt is [N][K] row-major
// (column-major view of the [K][N] source): WMMA B-fragment loads become
// contiguous 16-byte (8-half) reads.
// ---------------------------------------------------------------------------
__global__ void k_transpose_bf16(const float* __restrict__ W, bf16* __restrict__ Wt,
                                 int K, int N) {
  size_t idx = (size_t)blockIdx.x * blockDim.x + threadIdx.x;
  if (idx >= (size_t)K * N) return;
  int n = (int)(idx / K);
  int k = (int)(idx % K);
  Wt[idx] = (bf16)W[(size_t)k * N + n];
}

__global__ void k_bias_qkv(const float* __restrict__ bq, const float* __restrict__ bk,
                           const float* __restrict__ bv, float* __restrict__ bqkv) {
  int i = blockIdx.x * blockDim.x + threadIdx.x;
  if (i >= 3 * RD) return;
  bqkv[i] = (i < RD) ? bq[i] : (i < 2 * RD) ? bk[i - RD] : bv[i - 2 * RD];
}

__global__ void k_bias_gate(const float* __restrict__ bg, const float* __restrict__ bu,
                            float* __restrict__ bgu) {
  int i = blockIdx.x * blockDim.x + threadIdx.x;
  if (i >= 2 * RD) return;
  bgu[i] = bg[i] + bu[i];
}

// ---------------------------------------------------------------------------
// Per-step operand build: Akv[b][s][:] = bf16(mem) for s<3, bf16(x_t) for s==3;
// Atanh = bf16(tanh(mem)).
// ---------------------------------------------------------------------------
__global__ void k_build(const float* __restrict__ mem, const float* __restrict__ x,
                        bf16* __restrict__ Akv, bf16* __restrict__ Atanh, int t) {
  int idx = blockIdx.x * blockDim.x + threadIdx.x;  // < B*4*D
  int col = idx & (RD - 1);
  int r = idx >> 10;
  int b = r >> 2, s = r & 3;
  if (s < RS) {
    float m = mem[(size_t)(b * RS + s) * RD + col];
    Akv[idx] = (bf16)m;
    Atanh[(size_t)(b * RS + s) * RD + col] = (bf16)tanhf(m);
  } else {
    float xv = x[((size_t)b * RT + t) * RD + col];
    Akv[idx] = (bf16)xv;
  }
}

// ---------------------------------------------------------------------------
// bf16 WMMA GEMM:  C[M,N] = A[M,K] @ Wt[N,K]^T  (+bias[n]) (+res) (relu?)
//   A   : row-major bf16, leading dim lda (elements)
//   Wt  : row-major [N][K] bf16 (pre-transposed weights)
//   res : optional f32 residual, indexed res[(row/res_div)*resN + col]
//   Cf  : optional f32 output [M][N];  Cb: optional bf16 output [M][N]
// One wave computes a 32x32 C block as 2x2 WMMA tiles (4 v8f accumulators);
// A fragments reused across the 2 N-tiles, B fragments across the 2 M-tiles.
// Block = 256 threads = 8 waves = 8 N-blocks of 32.  grid = (N/256, M/32).
//
// Fragment layouts per CDNA5 ISA 7.12.2 (16-bit, wave32):
//   A: lanes 0-15 row m+l, K {kk..kk+7, kk+16..kk+23};
//      lanes 16-31 same rows, K {kk+8..kk+15, kk+24..kk+31}.
//   B: lanes 0-15 col n+l, K {kk..kk+7, kk+16..kk+23};
//      lanes 16-31 same cols, K {kk+8..kk+15, kk+24..kk+31}.
//   C/D: VGPR j -> row j (+8 for hi half-wave), col = lane&15.
// ---------------------------------------------------------------------------
__global__ void k_gemm_bf16(const bf16* __restrict__ A, int lda,
                            const bf16* __restrict__ Wt,
                            const float* __restrict__ bias,
                            const float* __restrict__ res, int res_div, int resN,
                            float* __restrict__ Cf, bf16* __restrict__ Cb,
                            int M, int N, int K, int relu) {
  const int lane = threadIdx.x & 31;
  const int wave = threadIdx.x >> 5;
  const int l15  = lane & 15;
  const int hi   = lane >> 4;                   // 0 or 1 (half-wave)
  const int n0   = (blockIdx.x * 8 + wave) * 32;
  const int m0   = blockIdx.y * 32;
  if (n0 >= N || m0 >= M) return;               // wave-uniform

  v8f c00 = {}, c01 = {}, c10 = {}, c11 = {};
  const bf16* arow0 = A  + (size_t)(m0 + l15) * lda;
  const bf16* arow1 = A  + (size_t)(m0 + 16 + l15) * lda;
  const bf16* brow0 = Wt + (size_t)(n0 + l15) * K;
  const bf16* brow1 = Wt + (size_t)(n0 + 16 + l15) * K;

  // Warm the lines for the first clause (global_prefetch_b8).
  __builtin_prefetch(arow0, 0, 3);
  __builtin_prefetch(arow1, 0, 3);
  __builtin_prefetch(brow0, 0, 3);
  __builtin_prefetch(brow1, 0, 3);

  for (int kk = 0; kk < K; kk += 32) {
    union { v16bf v; v8bf h[2]; } a0, a1, bu0, bu1;
    a0.h[0] = *(const v8bf*)(arow0 + kk + hi * 8);
    a0.h[1] = *(const v8bf*)(arow0 + kk + 16 + hi * 8);
    a1.h[0] = *(const v8bf*)(arow1 + kk + hi * 8);
    a1.h[1] = *(const v8bf*)(arow1 + kk + 16 + hi * 8);
    bu0.h[0] = *(const v8bf*)(brow0 + kk + hi * 8);
    bu0.h[1] = *(const v8bf*)(brow0 + kk + 16 + hi * 8);
    bu1.h[0] = *(const v8bf*)(brow1 + kk + hi * 8);
    bu1.h[1] = *(const v8bf*)(brow1 + kk + 16 + hi * 8);
    v16bf b0 = bu0.v;
    v16bf b1 = bu1.v;
    c00 = __builtin_amdgcn_wmma_f32_16x16x32_bf16(false, a0.v, false, b0, (short)0, c00, false, false);
    c01 = __builtin_amdgcn_wmma_f32_16x16x32_bf16(false, a0.v, false, b1, (short)0, c01, false, false);
    c10 = __builtin_amdgcn_wmma_f32_16x16x32_bf16(false, a1.v, false, b0, (short)0, c10, false, false);
    c11 = __builtin_amdgcn_wmma_f32_16x16x32_bf16(false, a1.v, false, b1, (short)0, c11, false, false);
  }

  const v8f* acc[4] = { &c00, &c01, &c10, &c11 };
  #pragma unroll
  for (int tm = 0; tm < 2; ++tm) {
    #pragma unroll
    for (int tn = 0; tn < 2; ++tn) {
      const v8f& c = *acc[tm * 2 + tn];
      const int col = n0 + tn * 16 + l15;
      #pragma unroll
      for (int j = 0; j < 8; ++j) {
        int row = m0 + tm * 16 + j + hi * 8;
        float v = c[j];
        if (bias) v += bias[col];
        if (res)  v += res[(size_t)(row / res_div) * resN + col];
        if (relu) v = fmaxf(v, 0.0f);
        if (Cf) Cf[(size_t)row * N + col] = v;
        if (Cb) Cb[(size_t)row * N + col] = (bf16)v;
      }
    }
  }
}

// ---------------------------------------------------------------------------
// Attention: one wave per (b,h); 4 d-elements per lane; dot products via
// __shfl_xor reductions (no LDS, no barriers). S=3 queries over 4 keys.
// QKV is [B*4][3072] f32: cols [0,1024)=Q, [1024,2048)=K, [2048,3072)=V.
// Writes O as bf16 [B*3][1024] (input to the Wo GEMM).
// ---------------------------------------------------------------------------
__global__ void k_attn(const float* __restrict__ QKV, bf16* __restrict__ O) {
  const int wave = threadIdx.x >> 5;
  const int lane = threadIdx.x & 31;
  const int bh = blockIdx.x * 8 + wave;         // < B*H = 512
  const int b = bh >> 3, h = bh & 7;
  const int d0 = lane * 4;

  float kj[4][4], vj[4][4];
  #pragma unroll
  for (int j = 0; j < 4; ++j) {
    const float* base = QKV + (size_t)(b * 4 + j) * (3 * RD) + h * RDK + d0;
    #pragma unroll
    for (int i = 0; i < 4; ++i) {
      kj[j][i] = base[RD + i];
      vj[j][i] = base[2 * RD + i];
    }
  }

  for (int s = 0; s < RS; ++s) {
    const float* qb = QKV + (size_t)(b * 4 + s) * (3 * RD) + h * RDK + d0;
    float q[4];
    #pragma unroll
    for (int i = 0; i < 4; ++i) q[i] = qb[i];

    float sc[4];
    #pragma unroll
    for (int j = 0; j < 4; ++j) {
      float p = q[0] * kj[j][0] + q[1] * kj[j][1] + q[2] * kj[j][2] + q[3] * kj[j][3];
      #pragma unroll
      for (int off = 16; off > 0; off >>= 1) p += __shfl_xor(p, off, 32);
      sc[j] = p * 0.08838834764831845f;          // 1/sqrt(128)
    }
    float mx = fmaxf(fmaxf(sc[0], sc[1]), fmaxf(sc[2], sc[3]));
    float e0 = expf(sc[0] - mx), e1 = expf(sc[1] - mx);
    float e2 = expf(sc[2] - mx), e3 = expf(sc[3] - mx);
    float inv = 1.0f / (e0 + e1 + e2 + e3);
    float p0 = e0 * inv, p1 = e1 * inv, p2 = e2 * inv, p3 = e3 * inv;

    bf16* ob = O + (size_t)(b * RS + s) * RD + h * RDK + d0;
    #pragma unroll
    for (int i = 0; i < 4; ++i) {
      float o = p0 * vj[0][i] + p1 * vj[1][i] + p2 * vj[2][i] + p3 * vj[3][i];
      ob[i] = (bf16)o;
    }
  }
}

// ---------------------------------------------------------------------------
// Gated update + output write:
//   nm2 = nm + ml;  new_mem = sigmoid(ig)*tanh(nm2) + sigmoid(fg)*mem
// ---------------------------------------------------------------------------
__global__ void k_final(const float* __restrict__ nm, const float* __restrict__ ml,
                        const float* __restrict__ G, float* __restrict__ mem,
                        float* __restrict__ out, int t) {
  int idx = blockIdx.x * blockDim.x + threadIdx.x;  // < 192*1024
  int col = idx & (RD - 1);
  int r = idx >> 10;            // 0..191
  int b = r / RS, s = r % RS;
  float nm2 = nm[idx] + ml[idx];
  float ig = G[(size_t)r * (2 * RD) + col];
  float fg = G[(size_t)r * (2 * RD) + RD + col];
  float o = (1.0f / (1.0f + expf(-ig))) * tanhf(nm2) +
            (1.0f / (1.0f + expf(-fg))) * mem[idx];
  mem[idx] = o;
  out[((size_t)b * RT + t) * (RS * RD) + (size_t)s * RD + col] = o;
}

// ---------------------------------------------------------------------------
extern "C" void kernel_launch(void* const* d_in, const int* in_sizes, int n_in,
                              void* d_out, int out_size, void* d_ws, size_t ws_size,
                              hipStream_t stream) {
  (void)in_sizes; (void)n_in; (void)out_size; (void)ws_size;
  const float* x    = (const float*)d_in[0];
  const float* mem0 = (const float*)d_in[1];
  const float* Wq_  = (const float*)d_in[2];  const float* bq_ = (const float*)d_in[3];
  const float* Wk_  = (const float*)d_in[4];  const float* bk_ = (const float*)d_in[5];
  const float* Wv_  = (const float*)d_in[6];  const float* bv_ = (const float*)d_in[7];
  const float* Wo_  = (const float*)d_in[8];  const float* bo_ = (const float*)d_in[9];
  const float* W1_  = (const float*)d_in[10]; const float* b1_ = (const float*)d_in[11];
  const float* W2_  = (const float*)d_in[12]; const float* b2_ = (const float*)d_in[13];
  const float* Wg_  = (const float*)d_in[14]; const float* bg_ = (const float*)d_in[15];
  const float* Ug_  = (const float*)d_in[16]; const float* bu_ = (const float*)d_in[17];
  float* out = (float*)d_out;

  // ---- workspace carve (256B aligned) ----
  char* p = (char*)d_ws;
  auto carve = [&](size_t bytes) -> char* {
    char* r = p; p += (bytes + 255) & ~(size_t)255; return r;
  };
  bf16*  Wqkvt = (bf16*)carve(3072ull * 1024 * 2);
  bf16*  Wot   = (bf16*)carve(1024ull * 1024 * 2);
  bf16*  W1t   = (bf16*)carve(1024ull * 1024 * 2);
  bf16*  W2t   = (bf16*)carve(1024ull * 1024 * 2);
  bf16*  Wgt   = (bf16*)carve(2048ull * 1024 * 2);
  bf16*  Ugt   = (bf16*)carve(2048ull * 1024 * 2);
  float* bqkv  = (float*)carve(3072ull * 4);
  float* bgu   = (float*)carve(2048ull * 4);
  float* memb  = (float*)carve(192ull * 1024 * 4);
  bf16*  Akv   = (bf16*)carve(256ull * 1024 * 2);
  bf16*  Atanh = (bf16*)carve(192ull * 1024 * 2);
  float* QKV   = (float*)carve(256ull * 3072 * 4);
  bf16*  Obf   = (bf16*)carve(192ull * 1024 * 2);
  float* Gx    = (float*)carve(64ull * 2048 * 4);
  float* G     = (float*)carve(192ull * 2048 * 4);
  float* nm_f  = (float*)carve(192ull * 1024 * 4);
  bf16*  nm_b  = (bf16*)carve(192ull * 1024 * 2);
  bf16*  h_b   = (bf16*)carve(192ull * 1024 * 2);
  float* ml_f  = (float*)carve(192ull * 1024 * 4);

  // ---- prep: state copy + weight transpose/convert (weights stay hot in L2)
  hipMemcpyAsync(memb, mem0, 192ull * 1024 * 4, hipMemcpyDeviceToDevice, stream);
  const int gT = (1024 * 1024) / 256;
  k_transpose_bf16<<<gT, 256, 0, stream>>>(Wq_, Wqkvt,                   1024, 1024);
  k_transpose_bf16<<<gT, 256, 0, stream>>>(Wk_, Wqkvt + 1024ull * 1024,  1024, 1024);
  k_transpose_bf16<<<gT, 256, 0, stream>>>(Wv_, Wqkvt + 2048ull * 1024,  1024, 1024);
  k_transpose_bf16<<<gT, 256, 0, stream>>>(Wo_, Wot, 1024, 1024);
  k_transpose_bf16<<<gT, 256, 0, stream>>>(W1_, W1t, 1024, 1024);
  k_transpose_bf16<<<gT, 256, 0, stream>>>(W2_, W2t, 1024, 1024);
  k_transpose_bf16<<<2 * gT, 256, 0, stream>>>(Wg_, Wgt, 1024, 2048);
  k_transpose_bf16<<<2 * gT, 256, 0, stream>>>(Ug_, Ugt, 1024, 2048);
  k_bias_qkv<<<12, 256, 0, stream>>>(bq_, bk_, bv_, bqkv);
  k_bias_gate<<<8, 256, 0, stream>>>(bg_, bu_, bgu);

  // ---- scan over T ----
  for (int t = 0; t < RT; ++t) {
    k_build<<<1024, 256, 0, stream>>>(memb, x, Akv, Atanh, t);
    // Q,K,V fused: [256,1024] @ [1024,3072]
    k_gemm_bf16<<<dim3(12, 8), 256, 0, stream>>>(
        Akv, 1024, Wqkvt, bqkv, nullptr, 1, 0, QKV, nullptr, 256, 3072, 1024, 0);
    // Gx = x @ Wg : [64,1024] @ [1024,2048]  (A = row 3 of each kv group)
    k_gemm_bf16<<<dim3(8, 2), 256, 0, stream>>>(
        Akv + 3 * 1024, 4096, Wgt, nullptr, nullptr, 1, 0, Gx, nullptr, 64, 2048, 1024, 0);
    // G = tanh(mem) @ Ug + (bg+bu) + broadcast Gx : [192,1024] @ [1024,2048]
    k_gemm_bf16<<<dim3(8, 6), 256, 0, stream>>>(
        Atanh, 1024, Ugt, bgu, Gx, 3, 2048, G, nullptr, 192, 2048, 1024, 0);
    // tiny attention: softmax over 4 keys per (b,h), one wave each
    k_attn<<<64, 256, 0, stream>>>(QKV, Obf);
    // nm = mem + O @ Wo + bo
    k_gemm_bf16<<<dim3(4, 6), 256, 0, stream>>>(
        Obf, 1024, Wot, bo_, memb, 1, 1024, nm_f, nm_b, 192, 1024, 1024, 0);
    // h = relu(nm @ W1 + b1)   (bf16 output only, feeds next GEMM)
    k_gemm_bf16<<<dim3(4, 6), 256, 0, stream>>>(
        nm_b, 1024, W1t, b1_, nullptr, 1, 0, nullptr, h_b, 192, 1024, 1024, 1);
    // ml = relu(h @ W2 + b2)
    k_gemm_bf16<<<dim3(4, 6), 256, 0, stream>>>(
        h_b, 1024, W2t, b2_, nullptr, 1, 0, ml_f, nullptr, 192, 1024, 1024, 1);
    // gated update + output write
    k_final<<<768, 256, 0, stream>>>(nm_f, ml_f, G, memb, out, t);
  }
}
